// Generator_23063974379766
// MI455X (gfx1250) — matrix operand order repaired and checked by reading, not compiled, not run on hardware
//
#include <hip/hip_runtime.h>
#include <hip/hip_bf16.h>

// ---------------------------------------------------------------------------
// Types for CDNA5 WMMA (wave32): v_wmma_f32_16x16x32_bf16
// ---------------------------------------------------------------------------
typedef __attribute__((ext_vector_type(8)))  float        v8f;
typedef __attribute__((ext_vector_type(8)))  unsigned int v8u;
typedef __attribute__((ext_vector_type(16))) __bf16       v16bf;

#define BB   8192
#define HH   512
#define HIN  128
#define NN   512
#define RW   16
#define K_LSTM 640   // HIN + HH

typedef unsigned short ushort_t;

__device__ __forceinline__ ushort_t f2bf(float f) {
  unsigned int u = __float_as_uint(f);
  u = (u + 0x7FFFu + ((u >> 16) & 1u)) >> 16;          // RNE
  return (ushort_t)u;
}

__device__ __forceinline__ float bf2f(ushort_t b) {
  return __uint_as_float((unsigned int)b << 16);
}

__device__ __forceinline__ float sigm(float x) { return 1.0f / (1.0f + expf(-x)); }

// A fragment (16x32 bf16). ISA dense-A layout: lanes 0-15 hold K runs
// {0..7, 16..23}; lanes 16-31 hold {8..15, 24..31}. p is the per-lane base
// (row + half-select already applied): two 16B runs at +0 and +16 elements.
__device__ __forceinline__ v8u ldA16(const ushort_t* __restrict__ p) {
  const uint4 q0 = *(const uint4*)(p);
  const uint4 q1 = *(const uint4*)(p + 16);
  v8u f;
  f[0] = q0.x; f[1] = q0.y; f[2] = q0.z; f[3] = q0.w;
  f[4] = q1.x; f[5] = q1.y; f[6] = q1.z; f[7] = q1.w;
  return f;
}

// B fragment (32x16 bf16): lanes 0-15 hold K runs {0..7, 16..23}, lanes 16-31
// {8..15, 24..31}; two 16B runs at +0 and +16 elements (half-select applied).
__device__ __forceinline__ v8u ldB16(const ushort_t* __restrict__ p) {
  const uint4 q0 = *(const uint4*)(p);
  const uint4 q1 = *(const uint4*)(p + 16);
  v8u f;
  f[0] = q0.x; f[1] = q0.y; f[2] = q0.z; f[3] = q0.w;
  f[4] = q1.x; f[5] = q1.y; f[6] = q1.z; f[7] = q1.w;
  return f;
}

__device__ __forceinline__ v8f wmma_bf16(v8u a, v8u b, v8f c) {
  return __builtin_amdgcn_wmma_f32_16x16x32_bf16(
      false, __builtin_bit_cast(v16bf, a),
      false, __builtin_bit_cast(v16bf, b),
      (short)0, c, false, false);
}

// ---------------------------------------------------------------------------
// Weight transpose + fp32 -> bf16 convert:  Wt[n*K + k] = bf16(W[k*N + n])
// ---------------------------------------------------------------------------
__global__ void transpose_bf16_kernel(const float* __restrict__ W,
                                      ushort_t* __restrict__ Wt,
                                      int K, int N) {
  int idx = blockIdx.x * blockDim.x + threadIdx.x;
  if (idx >= K * N) return;
  int k = idx / N, n = idx % N;
  Wt[(size_t)n * K + k] = f2bf(W[idx]);
}

// Plain fp32 -> bf16 convert (same layout)
__global__ void convert_bf16_kernel(const float* __restrict__ W,
                                    ushort_t* __restrict__ Wb, int n) {
  int idx = blockIdx.x * blockDim.x + threadIdx.x;
  if (idx < n) Wb[idx] = f2bf(W[idx]);
}

// ---------------------------------------------------------------------------
// Generic GEMM: C[M,512] = act(A[M,K] @ Wt^T + bias).  WG 256 = 8 waves,
// WG tile 128x64, wave tile 32x32 (2x2 WMMA frags).  K is compile-time so the
// unrolled k-steps fold into global_load immediate offsets (no in-loop VALU
// beyond 4 pointer bumps per 128 K-elements).
// ACT: 0=none, 1=tanh.  OBF: 0=fp32 out, 1=bf16 out.
// ---------------------------------------------------------------------------
template <int ACT, int OBF, int K>
__global__ __launch_bounds__(256)
void gemm_kernel(const ushort_t* __restrict__ A,
                 const ushort_t* __restrict__ Wt,
                 const float* __restrict__ bias,
                 void* __restrict__ Cv) {
  int lane = threadIdx.x & 31;
  int wave = threadIdx.x >> 5;
  int m0 = blockIdx.y * 128 + (wave & 3) * 32;
  int n0 = blockIdx.x * 64  + (wave >> 2) * 32;
  int ha = (lane >> 4) << 3;                           // 0 or 8
  int hb = (lane >> 4) << 3;                           // 0 or 8
  const ushort_t* pA0 = A  + (size_t)(m0 + (lane & 15)) * K + ha;
  const ushort_t* pA1 = pA0 + (size_t)16 * K;
  const ushort_t* pB0 = Wt + (size_t)(n0 + (lane & 15)) * K + hb;
  const ushort_t* pB1 = pB0 + (size_t)16 * K;

  v8f acc[2][2] = {};
  for (int ko = 0; ko < K; ko += 128) {
#pragma unroll
    for (int ku = 0; ku < 128; ku += 32) {
      v8u a0 = ldA16(pA0 + ku);
      v8u a1 = ldA16(pA1 + ku);
      v8u b0 = ldB16(pB0 + ku);
      v8u b1 = ldB16(pB1 + ku);
      acc[0][0] = wmma_bf16(a0, b0, acc[0][0]);
      acc[0][1] = wmma_bf16(a0, b1, acc[0][1]);
      acc[1][0] = wmma_bf16(a1, b0, acc[1][0]);
      acc[1][1] = wmma_bf16(a1, b1, acc[1][1]);
    }
    pA0 += 128; pA1 += 128; pB0 += 128; pB1 += 128;
  }

  int cm = (lane >> 4) << 3;                           // 0 or 8
  int cn = lane & 15;
#pragma unroll
  for (int i = 0; i < 2; ++i)
#pragma unroll
    for (int j = 0; j < 2; ++j) {
      int ccol = n0 + j * 16 + cn;
      float bv = bias[ccol];
#pragma unroll
      for (int v = 0; v < 8; ++v) {
        int crow = m0 + i * 16 + cm + v;
        float val = acc[i][j][v] + bv;
        if (ACT) val = tanhf(val);
        if (OBF) ((ushort_t*)Cv)[(size_t)crow * NN + ccol] = f2bf(val);
        else     ((float*)Cv)   [(size_t)crow * NN + ccol] = val;
      }
    }
}

// ---------------------------------------------------------------------------
// Fused LSTM step: gates = [X|Hp] @ W_lstm + b (K = 128 + 512, Nout = 2048),
// each wave computes its 32x32 (m,n) patch for ALL FOUR gate blocks; the
// LSTM nonlinearity fuses into the epilogue (no gates buffer).
//   cy = cprev * sigm(f + 1) + sigm(i) * tanh(g);  hy = sigm(o) * tanh(cy)
// X, Hp bf16; Cprev/Cout fp32 (alias OK); Hout written bf16.  Gate offsets
// (g*HH*K_LSTM elements, <= 1.97MB) fold into the 24-bit load immediates.
// ---------------------------------------------------------------------------
__global__ __launch_bounds__(256)
void lstm_gemm_kernel(const ushort_t* __restrict__ X,       // [M, HIN] bf16
                      const ushort_t* __restrict__ Hp,      // [M, HH]  bf16
                      const ushort_t* __restrict__ Wt,      // [4*HH, K_LSTM] bf16
                      const float* __restrict__ bias,       // [4*HH]
                      const float* __restrict__ Cprev,      // [M, HH] fp32
                      ushort_t* __restrict__ Hout,          // [M, HH] bf16
                      float* __restrict__ Cout) {           // may alias Cprev
  int lane = threadIdx.x & 31;
  int wave = threadIdx.x >> 5;
  int m0 = blockIdx.y * 128 + (wave & 3) * 32;
  int n0 = blockIdx.x * 64  + (wave >> 2) * 32;        // n in [0, HH)
  int ha = (lane >> 4) << 3;
  int hb = (lane >> 4) << 3;
  v8f acc[4][2][2] = {};

  // K-segment 1: X (K = HIN = 128, weight cols 0..HIN) -- fully unrolled
  {
    const ushort_t* pA0 = X  + (size_t)(m0 + (lane & 15)) * HIN + ha;
    const ushort_t* pA1 = pA0 + (size_t)16 * HIN;
    const ushort_t* pB0 = Wt + (size_t)(n0 + (lane & 15)) * K_LSTM + hb;
    const ushort_t* pB1 = pB0 + (size_t)16 * K_LSTM;
#pragma unroll
    for (int ku = 0; ku < HIN; ku += 32) {
      v8u a0 = ldA16(pA0 + ku);
      v8u a1 = ldA16(pA1 + ku);
#pragma unroll
      for (int g = 0; g < 4; ++g) {
        v8u b0 = ldB16(pB0 + (size_t)g * HH * K_LSTM + ku);
        v8u b1 = ldB16(pB1 + (size_t)g * HH * K_LSTM + ku);
        acc[g][0][0] = wmma_bf16(a0, b0, acc[g][0][0]);
        acc[g][0][1] = wmma_bf16(a0, b1, acc[g][0][1]);
        acc[g][1][0] = wmma_bf16(a1, b0, acc[g][1][0]);
        acc[g][1][1] = wmma_bf16(a1, b1, acc[g][1][1]);
      }
    }
  }
  // K-segment 2: Hp (K = HH = 512, weight cols HIN..HIN+HH), unroll x4
  {
    const ushort_t* pA0 = Hp + (size_t)(m0 + (lane & 15)) * HH + ha;
    const ushort_t* pA1 = pA0 + (size_t)16 * HH;
    const ushort_t* pB0 = Wt + (size_t)(n0 + (lane & 15)) * K_LSTM + hb + HIN;
    const ushort_t* pB1 = pB0 + (size_t)16 * K_LSTM;
    for (int ko = 0; ko < HH; ko += 128) {
#pragma unroll
      for (int ku = 0; ku < 128; ku += 32) {
        v8u a0 = ldA16(pA0 + ku);
        v8u a1 = ldA16(pA1 + ku);
#pragma unroll
        for (int g = 0; g < 4; ++g) {
          v8u b0 = ldB16(pB0 + (size_t)g * HH * K_LSTM + ku);
          v8u b1 = ldB16(pB1 + (size_t)g * HH * K_LSTM + ku);
          acc[g][0][0] = wmma_bf16(a0, b0, acc[g][0][0]);
          acc[g][0][1] = wmma_bf16(a0, b1, acc[g][0][1]);
          acc[g][1][0] = wmma_bf16(a1, b0, acc[g][1][0]);
          acc[g][1][1] = wmma_bf16(a1, b1, acc[g][1][1]);
        }
      }
      pA0 += 128; pA1 += 128; pB0 += 128; pB1 += 128;
    }
  }

  int cm = (lane >> 4) << 3;
  int cn = lane & 15;
#pragma unroll
  for (int i = 0; i < 2; ++i)
#pragma unroll
    for (int j = 0; j < 2; ++j) {
      int ccol = n0 + j * 16 + cn;
      float bi = bias[ccol];
      float bg = bias[HH + ccol];
      float bf = bias[2 * HH + ccol];
      float bo = bias[3 * HH + ccol];
#pragma unroll
      for (int v = 0; v < 8; ++v) {
        int crow = m0 + i * 16 + cm + v;
        size_t off = (size_t)crow * HH + ccol;
        float iv = acc[0][i][j][v] + bi;
        float gv = acc[1][i][j][v] + bg;
        float fv = acc[2][i][j][v] + bf;
        float ov = acc[3][i][j][v] + bo;
        float cy = Cprev[off] * sigm(fv + 1.0f) + sigm(iv) * tanhf(gv);
        float hy = sigm(ov) * tanhf(cy);
        Cout[off] = cy;
        Hout[off] = f2bf(hy);
      }
    }
}

// ---------------------------------------------------------------------------
// Gumbel sampling: straight-through output is exactly one-hot(argmax(logits+g)).
// One block per row b: argmax reduce, write one-hot row of out_steps, and
// gather bf16 rows: x_next = Wd_bf[amax], xl = Wdl_bf[amax].
// ---------------------------------------------------------------------------
__global__ __launch_bounds__(256)
void sample_kernel(const float* __restrict__ logits,
                   const float* __restrict__ gu,         // gumbel_u + t*B*N
                   const ushort_t* __restrict__ Wd_bf,   // [NN, HIN] bf16
                   const ushort_t* __restrict__ Wdl_bf,  // [NN, HIN] bf16
                   float* __restrict__ out1,             // d_out, [B, RW, NN]
                   int t,
                   ushort_t* __restrict__ Xn,            // [B, HIN] bf16
                   ushort_t* __restrict__ Xl) {          // [B, HIN] bf16
  __shared__ float smax[256];
  __shared__ int   sidx[256];
  int b = blockIdx.x;
  int tid = threadIdx.x;
  float best = -3.0e38f;
  int bidx = 0;
  for (int n = tid; n < NN; n += 256) {
    float u = gu[(size_t)b * NN + n];
    float g = -logf(-logf(u + 1e-20f) + 1e-20f);
    float v = logits[(size_t)b * NN + n] + g;
    if (v > best) { best = v; bidx = n; }
  }
  smax[tid] = best; sidx[tid] = bidx;
  __syncthreads();
  for (int s = 128; s > 0; s >>= 1) {
    if (tid < s && smax[tid + s] > smax[tid]) {
      smax[tid] = smax[tid + s]; sidx[tid] = sidx[tid + s];
    }
    __syncthreads();
  }
  int amax = sidx[0];
  float* orow = out1 + (size_t)b * RW * NN + (size_t)t * NN;
  for (int n = tid; n < NN; n += 256) orow[n] = (n == amax) ? 1.0f : 0.0f;
  if (tid < HIN) {
    Xn[(size_t)b * HIN + tid] = Wd_bf [(size_t)amax * HIN + tid];
    Xl[(size_t)b * HIN + tid] = Wdl_bf[(size_t)amax * HIN + tid];
  }
}

// ---------------------------------------------------------------------------
// line = hl @ W_outl + b_outl : one wave (32 lanes) per row, shuffle reduce.
// Hl is bf16.
// ---------------------------------------------------------------------------
__global__ __launch_bounds__(256)
void outl_kernel(const ushort_t* __restrict__ Hl,
                 const float* __restrict__ W_outl,
                 const float* __restrict__ b_outl, float* __restrict__ out2,
                 int t) {
  int lane = threadIdx.x & 31;
  int wave = threadIdx.x >> 5;
  int b = blockIdx.x * 8 + wave;
  const ushort_t* h = Hl + (size_t)b * HH;
  float s = 0.0f;
#pragma unroll
  for (int j = 0; j < 16; ++j)
    s += bf2f(h[lane + 32 * j]) * W_outl[lane + 32 * j];
#pragma unroll
  for (int off = 16; off > 0; off >>= 1) s += __shfl_xor(s, off, 32);
  if (lane == 0) out2[(size_t)b * RW + t] = s + b_outl[0];
}

// ---------------------------------------------------------------------------
// Host launcher
// ---------------------------------------------------------------------------
extern "C" void kernel_launch(void* const* d_in, const int* in_sizes, int n_in,
                              void* d_out, int out_size, void* d_ws, size_t ws_size,
                              hipStream_t stream) {
  const float* latent   = (const float*)d_in[0];
  const float* inputs   = (const float*)d_in[1];
  const float* gumbel_u = (const float*)d_in[2];
  const float* W_int  = (const float*)d_in[3];  const float* b_int  = (const float*)d_in[4];
  const float* W_intl = (const float*)d_in[5];  const float* b_intl = (const float*)d_in[6];
  const float* W_hup  = (const float*)d_in[7];  const float* b_hup  = (const float*)d_in[8];
  const float* W_cup  = (const float*)d_in[9];  const float* b_cup  = (const float*)d_in[10];
  const float* W_hupl = (const float*)d_in[11]; const float* b_hupl = (const float*)d_in[12];
  const float* W_cupl = (const float*)d_in[13]; const float* b_cupl = (const float*)d_in[14];
  const float* W_lstm = (const float*)d_in[15]; const float* b_lstm = (const float*)d_in[16];
  const float* W_up   = (const float*)d_in[17]; const float* b_up   = (const float*)d_in[18];
  const float* W_down  = (const float*)d_in[19];
  const float* W_downl = (const float*)d_in[20];
  const float* W_outl  = (const float*)d_in[21];
  const float* b_outl  = (const float*)d_in[22];

  float* out1 = (float*)d_out;                               // [B, RW, NN]
  float* out2 = (float*)d_out + (size_t)BB * RW * NN;        // [B, RW, 1]

  // Workspace carve-out (256B aligned)
  char* ws = (char*)d_ws;
  size_t off = 0;
  auto carve = [&](size_t bytes) -> char* {
    char* p = ws + off;
    off += (bytes + 255) & ~(size_t)255;
    return p;
  };
  ushort_t* wt_lstm = (ushort_t*)carve((size_t)4 * HH * K_LSTM * 2);
  ushort_t* wt_up   = (ushort_t*)carve((size_t)NN * HH * 2);
  ushort_t* wt_int  = (ushort_t*)carve((size_t)HH * HIN * 2);
  ushort_t* wt_intl = (ushort_t*)carve((size_t)HH * HIN * 2);
  ushort_t* wt_hup  = (ushort_t*)carve((size_t)HH * HH * 2);
  ushort_t* wt_cup  = (ushort_t*)carve((size_t)HH * HH * 2);
  ushort_t* wt_hupl = (ushort_t*)carve((size_t)HH * HH * 2);
  ushort_t* wt_cupl = (ushort_t*)carve((size_t)HH * HH * 2);
  ushort_t* wd_bf   = (ushort_t*)carve((size_t)NN * HIN * 2);
  ushort_t* wdl_bf  = (ushort_t*)carve((size_t)NN * HIN * 2);
  float*    logits  = (float*)carve((size_t)BB * HH * 4);
  ushort_t* interbf = (ushort_t*)carve((size_t)BB * HH * 2);
  ushort_t* h0bf    = (ushort_t*)carve((size_t)BB * HH * 2);
  ushort_t* h1bf    = (ushort_t*)carve((size_t)BB * HH * 2);
  ushort_t* hl0bf   = (ushort_t*)carve((size_t)BB * HH * 2);
  ushort_t* hl1bf   = (ushort_t*)carve((size_t)BB * HH * 2);
  float*    cbuf    = (float*)carve((size_t)BB * HH * 4);
  float*    clbuf   = (float*)carve((size_t)BB * HH * 4);
  ushort_t* latbf   = (ushort_t*)carve((size_t)BB * HIN * 2);
  ushort_t* inbf    = (ushort_t*)carve((size_t)BB * HIN * 2);
  ushort_t* xbf     = (ushort_t*)carve((size_t)BB * HIN * 2);
  ushort_t* xlbf    = (ushort_t*)carve((size_t)BB * HIN * 2);

  // 1) Weight transpose / convert to bf16
  auto tr = [&](const float* W, ushort_t* Wt, int K, int N) {
    int n = K * N;
    transpose_bf16_kernel<<<(n + 255) / 256, 256, 0, stream>>>(W, Wt, K, N);
  };
  auto cv = [&](const float* W, ushort_t* Wb, int n) {
    convert_bf16_kernel<<<(n + 255) / 256, 256, 0, stream>>>(W, Wb, n);
  };
  tr(W_lstm, wt_lstm, K_LSTM, 4 * HH);
  tr(W_up,   wt_up,   HH, NN);
  tr(W_int,  wt_int,  HIN, HH);
  tr(W_intl, wt_intl, HIN, HH);
  tr(W_hup,  wt_hup,  HH, HH);
  tr(W_cup,  wt_cup,  HH, HH);
  tr(W_hupl, wt_hupl, HH, HH);
  tr(W_cupl, wt_cupl, HH, HH);
  cv(W_down,  wd_bf,  NN * HIN);
  cv(W_downl, wdl_bf, NN * HIN);
  cv(latent,  latbf,  BB * HIN);
  cv(inputs,  inbf,   BB * HIN);

  // 2) Initial states: inter = tanh(latent@W_int+b); hx/cx = tanh(inter@W_*up+b)
  dim3 gemm_grid(HH / 64, BB / 128);
  gemm_kernel<1,1,HIN><<<gemm_grid, 256, 0, stream>>>(latbf,   wt_int,  b_int,  interbf);
  gemm_kernel<1,1,HH ><<<gemm_grid, 256, 0, stream>>>(interbf, wt_hup,  b_hup,  h0bf);
  gemm_kernel<1,0,HH ><<<gemm_grid, 256, 0, stream>>>(interbf, wt_cup,  b_cup,  cbuf);
  gemm_kernel<1,1,HIN><<<gemm_grid, 256, 0, stream>>>(latbf,   wt_intl, b_intl, interbf);
  gemm_kernel<1,1,HH ><<<gemm_grid, 256, 0, stream>>>(interbf, wt_hupl, b_hupl, hl0bf);
  gemm_kernel<1,0,HH ><<<gemm_grid, 256, 0, stream>>>(interbf, wt_cupl, b_cupl, clbuf);

  // 3) 16 scan steps (both scans interleaved; scan-2 step t depends only on
  //    the one-hot sample from scan-1 step t)
  const ushort_t* curx = inbf;
  ushort_t* hcur = h0bf;  ushort_t* hnxt = h1bf;
  ushort_t* hlcur = hl0bf; ushort_t* hlnxt = hl1bf;
  for (int t = 0; t < RW; ++t) {
    lstm_gemm_kernel<<<gemm_grid, 256, 0, stream>>>(
        curx, hcur, wt_lstm, b_lstm, cbuf, hnxt, cbuf);
    gemm_kernel<0,0,HH><<<gemm_grid, 256, 0, stream>>>(
        hnxt, wt_up, b_up, logits);
    sample_kernel<<<BB, 256, 0, stream>>>(
        logits, gumbel_u + (size_t)t * BB * NN, wd_bf, wdl_bf,
        out1, t, xbf, xlbf);
    lstm_gemm_kernel<<<gemm_grid, 256, 0, stream>>>(
        xlbf, hlcur, wt_lstm, b_lstm, clbuf, hlnxt, clbuf);
    outl_kernel<<<BB / 8, 256, 0, stream>>>(hlnxt, W_outl, b_outl, out2, t);

    curx = xbf;
    ushort_t* tmp = hcur; hcur = hnxt; hnxt = tmp;
    tmp = hlcur; hlcur = hlnxt; hlnxt = tmp;
  }
}
